// DimeNet_39917426049349
// MI455X (gfx1250) — matrix operand order repaired and checked by reading, not compiled, not run on hardware
//
#include <hip/hip_runtime.h>
#include <hip/hip_bf16.h>
#include <math.h>

// ---------------------------------------------------------------------------
// Types for CDNA5 WMMA (gfx1250, wave32)
// ---------------------------------------------------------------------------
typedef __bf16 bf16_t;
typedef __attribute__((ext_vector_type(16))) __bf16 v16bf;
typedef __attribute__((ext_vector_type(8)))  float  v8f;
typedef __attribute__((ext_vector_type(4)))  float  v4f;

// DimeNet constants (match reference)
#define NS 7
#define NR 6
#define NBIL 8
#define NTGT 12

__device__ inline v8f wmma_bf16(v16bf a, v16bf b, v8f c) {
  // v_wmma_f32_16x16x32_bf16
  return __builtin_amdgcn_wmma_f32_16x16x32_bf16(false, a, false, b, (short)0, c,
                                                 false, false);
}

// A-matrix 16x32 bf16 lane layout: lane L -> row = L&15, k0 = (L>>4)*8;
// 16 values = K [k0..k0+7] then [k0+16..k0+23] (ISA 7.12.2).
__device__ inline v16bf load_a_tile(const float* __restrict__ rp, int k0) {
  v4f a0 = *(const v4f*)(rp + k0);
  v4f a1 = *(const v4f*)(rp + k0 + 4);
  v4f a2 = *(const v4f*)(rp + k0 + 16);
  v4f a3 = *(const v4f*)(rp + k0 + 20);
  v16bf r;
#pragma unroll
  for (int i = 0; i < 4; ++i) {
    r[i]      = (__bf16)a0[i];
    r[4 + i]  = (__bf16)a1[i];
    r[8 + i]  = (__bf16)a2[i];
    r[12 + i] = (__bf16)a3[i];
  }
  return r;
}

__device__ inline float swishf(float x) { return x / (1.0f + __expf(-x)); }

// ---------------------------------------------------------------------------
// Generic WMMA GEMM: out[M,ncols] = act(concat_seg(A)[M, NSEG*128] @ W + bias)
// optionally * mulsrc, + addsrc.  One wave computes a 16-row x (NTILES*16) tile.
// A rows are gathered f32 and converted to bf16 on the fly; W is pre-packed
// bf16 in exact WMMA B-operand lane order.
// ---------------------------------------------------------------------------
template<int NSEG, int NTILES>
__global__ __launch_bounds__(128) void wmma_gemm_kernel(
    const float* __restrict__ A0, const int* __restrict__ idx0,
    const float* __restrict__ A1, const int* __restrict__ idx1,
    const float* __restrict__ A2, const int* __restrict__ idx2,
    const bf16_t* __restrict__ Wp,
    const float* __restrict__ bias,
    const float* __restrict__ mulsrc,
    const float* __restrict__ addsrc,
    float* __restrict__ out,
    int M, int ncols, int out_stride, int act)
{
  constexpr int KSTEPS = NSEG * 4;
  const int lane  = threadIdx.x & 31;
  const int wave  = blockIdx.x * (blockDim.x >> 5) + (threadIdx.x >> 5);
  const int base  = wave * 16;
  if (base >= M) return;                 // wave-uniform: EXEC stays all-ones
  const int mlane = lane & 15;
  const int khalf = lane >> 4;
  int rowA = base + mlane;
  if (rowA >= M) rowA = M - 1;           // clamp (duplicate loads, stores guarded)

  const float* rp[NSEG];
  rp[0] = A0 + (size_t)(idx0 ? idx0[rowA] : rowA) * 128;
  if (NSEG > 1) rp[1] = A1 + (size_t)(idx1 ? idx1[rowA] : rowA) * 128;
  if (NSEG > 2) rp[2] = A2 + (size_t)(idx2 ? idx2[rowA] : rowA) * 128;

  v16bf a[KSTEPS];
#pragma unroll
  for (int kt = 0; kt < KSTEPS; ++kt)
    a[kt] = load_a_tile(rp[kt >> 2], (kt & 3) * 32 + khalf * 8);

#pragma unroll
  for (int nt = 0; nt < NTILES; ++nt) {
    v8f c = {};
    const bf16_t* wb = Wp + ((size_t)nt * KSTEPS) * 512 + (size_t)lane * 16;
#pragma unroll
    for (int kt = 0; kt < KSTEPS; ++kt) {
      v16bf b = *(const v16bf*)(wb + (size_t)kt * 512);
      c = wmma_bf16(a[kt], b, c);
    }
    const int col = nt * 16 + mlane;
    float bv = 0.0f;
    if (bias && col < ncols) bv = bias[col];
#pragma unroll
    for (int v = 0; v < 8; ++v) {
      const int rowO = base + v + 8 * khalf;   // C layout: vgpr v, lane half
      float val = c[v] + bv;
      if (act) val = swishf(val);
      if (rowO < M && col < ncols) {
        const size_t off = (size_t)rowO * out_stride + col;
        if (mulsrc) val *= mulsrc[off];
        if (addsrc) val += addsrc[off];
        out[off] = val;
      }
    }
  }
}

// ---------------------------------------------------------------------------
// Bilinear triplet kernel:
//   acc[t,:] = sum_j sb[t,j] * (kj[id_expand[t],:] @ M_j),  M_j[k][n]=Wb[n,j,k]
//   m[id_reduce[t],:] += acc[t,:]    (atomic f32 scatter)
// One wave per 16 triplets; A tile held in VGPRs across all 8 j-matrices:
// 8 j * 8 ntiles * 4 ksteps = 256 WMMAs per tile.
// ---------------------------------------------------------------------------
__global__ __launch_bounds__(128) void bilinear_kernel(
    const float* __restrict__ kjbuf, const int* __restrict__ id_expand,
    const float* __restrict__ sb,           // [T, 8]
    const bf16_t* __restrict__ Wb,          // 8 packed matrices, 16384 bf16 each
    const int* __restrict__ id_reduce,
    float* __restrict__ m,                  // [E,128] atomic accumulate
    int T)
{
  const int lane  = threadIdx.x & 31;
  const int wave  = blockIdx.x * (blockDim.x >> 5) + (threadIdx.x >> 5);
  const int base  = wave * 16;
  if (base >= T) return;
  const int mlane = lane & 15;
  const int khalf = lane >> 4;
  int rowT = base + mlane;
  if (rowT >= T) rowT = T - 1;
  const float* rp = kjbuf + (size_t)id_expand[rowT] * 128;

  v16bf a[4];
#pragma unroll
  for (int kt = 0; kt < 4; ++kt)
    a[kt] = load_a_tile(rp, kt * 32 + khalf * 8);

  int  rowsv[8], edges[8];
  bool validv[8];
#pragma unroll
  for (int v = 0; v < 8; ++v) {
    int r = base + v + 8 * khalf;
    bool ok = (r < T);
    int rc = ok ? r : (T - 1);
    rowsv[v]  = rc;
    validv[v] = ok;
    edges[v]  = id_reduce[rc];
  }

  float acc[8][8];
#pragma unroll
  for (int nt = 0; nt < 8; ++nt)
#pragma unroll
    for (int v = 0; v < 8; ++v) acc[nt][v] = 0.0f;

  for (int j = 0; j < NBIL; ++j) {
    float s[8];
#pragma unroll
    for (int v = 0; v < 8; ++v) s[v] = sb[(size_t)rowsv[v] * 8 + j];
    const bf16_t* wj = Wb + (size_t)j * (8 * 4 * 512);
#pragma unroll
    for (int nt = 0; nt < 8; ++nt) {
      v8f c = {};
      const bf16_t* wb = wj + (size_t)nt * 4 * 512 + (size_t)lane * 16;
#pragma unroll
      for (int kt = 0; kt < 4; ++kt) {
        v16bf b = *(const v16bf*)(wb + (size_t)kt * 512);
        c = wmma_bf16(a[kt], b, c);
      }
#pragma unroll
      for (int v = 0; v < 8; ++v) acc[nt][v] += s[v] * c[v];
    }
  }

#pragma unroll
  for (int nt = 0; nt < 8; ++nt) {
    const int col = nt * 16 + mlane;
#pragma unroll
    for (int v = 0; v < 8; ++v) {
      if (validv[v])
        atomicAdd(&m[(size_t)edges[v] * 128 + col], acc[nt][v]);
    }
  }
}

// ---------------------------------------------------------------------------
// Weight packing into WMMA B-operand order (bf16).
// B 32x16 tile: lane L -> n = L&15, k = kt*32 + (L>>4)*8 + (i&7) + (i>>3)*16.
// Generic stride addressing handles W[K][N], emb [384][128], W_bilin slices,
// and zero-pads final_W to N=16.
// ---------------------------------------------------------------------------
__global__ void pack_kernel(const float* __restrict__ W, bf16_t* __restrict__ dst,
                            int ksteps, int ntiles, int ncols,
                            long sK, long sN, long off)
{
  const long total = (long)ksteps * ntiles * 512;
  long gid = (long)blockIdx.x * blockDim.x + threadIdx.x;
  if (gid >= total) return;
  const int i    = (int)(gid & 15);
  const int lane = (int)((gid >> 4) & 31);
  const long rest = gid >> 9;
  const int kt = (int)(rest % ksteps);
  const int nt = (int)(rest / ksteps);
  const int n = nt * 16 + (lane & 15);
  const int k = kt * 32 + (lane >> 4) * 8 + (i & 7) + (i >> 3) * 16;
  float val = (n < ncols) ? W[(long)k * sK + (long)n * sN + off] : 0.0f;
  dst[gid] = (bf16_t)val;
}

// ---------------------------------------------------------------------------
// Spherical-Bessel zeros + norms (module-level numpy constants in reference),
// recomputed per launch by one thread in fp64 bisection (same algorithm).
// ---------------------------------------------------------------------------
__device__ double dsph_jn(int n, double x) {
  double j0 = sin(x) / x;
  if (n == 0) return j0;
  double jm = j0;
  double jc = sin(x) / (x * x) - cos(x) / x;
  for (int l = 1; l < n; ++l) {
    double nx = (2.0 * l + 1.0) / x * jc - jm;
    jm = jc; jc = nx;
  }
  return jc;
}

__global__ void basis_consts_kernel(float* __restrict__ Zs, float* __restrict__ Norm)
{
  if (threadIdx.x != 0 || blockIdx.x != 0) return;
  const double PI = 3.14159265358979323846;
  double zeros[NS][NR];
  double points[NS + NR - 1];
  double tmp[NS + NR - 1];
  for (int k = 0; k < NR; ++k) zeros[0][k] = (k + 1) * PI;
  int npts = NS + NR - 1;                       // 12
  for (int i = 0; i < npts; ++i) points[i] = (i + 1) * PI;
  for (int order = 1; order < NS; ++order) {
    for (int j = 0; j < npts - 1; ++j) {
      double a = points[j], b = points[j + 1];
      double fa = dsph_jn(order, a);
      for (int it = 0; it < 100; ++it) {
        double mid = 0.5 * (a + b);
        double fm = dsph_jn(order, mid);
        if (fa * fm <= 0.0) { b = mid; } else { a = mid; fa = fm; }
      }
      tmp[j] = 0.5 * (a + b);
    }
    npts -= 1;
    for (int j = 0; j < npts; ++j) points[j] = tmp[j];
    for (int k = 0; k < NR; ++k) zeros[order][k] = points[k];
  }
  for (int l = 0; l < NS; ++l)
    for (int r = 0; r < NR; ++r) {
      double z = zeros[l][r];
      double jl1 = dsph_jn(l + 1, z);
      Zs[l * NR + r]   = (float)z;
      Norm[l * NR + r] = (float)(1.0 / sqrt(0.5 * jl1 * jl1));
    }
}

// ---------------------------------------------------------------------------
// Per-edge geometry + radial bases:  rbf[E,6], srbf[E,42]
// ---------------------------------------------------------------------------
__global__ void edge_geom_kernel(const float* __restrict__ R,
                                 const int* __restrict__ nb_i,
                                 const int* __restrict__ nb_j,
                                 const float* __restrict__ freq,
                                 const float* __restrict__ Zs,
                                 const float* __restrict__ Norm,
                                 float* __restrict__ rbf,
                                 float* __restrict__ srbf, int E)
{
  int e = blockIdx.x * blockDim.x + threadIdx.x;
  if (e >= E) return;
  const int ia = nb_i[e], ja = nb_j[e];
  float dx = R[ia * 3 + 0] - R[ja * 3 + 0];
  float dy = R[ia * 3 + 1] - R[ja * 3 + 1];
  float dz = R[ia * 3 + 2] - R[ja * 3 + 2];
  float s = fmaxf(dx * dx + dy * dy + dz * dz, 0.0f);
  float d = sqrtf(s) * 0.2f;                    // / CUTOFF
  // envelope, p = 6: 1/x - 28 x^5 + 48 x^6 - 21 x^7  (0 for x >= 1)
  float x2 = d * d;
  float x5 = x2 * x2 * d;
  float env = 1.0f / d + x5 * (-28.0f + d * (48.0f - 21.0f * d));
  if (d >= 1.0f) env = 0.0f;
#pragma unroll
  for (int r = 0; r < NR; ++r)
    rbf[(size_t)e * NR + r] = env * __sinf(freq[r] * d);
  for (int l = 0; l < NS; ++l) {
#pragma unroll
    for (int r = 0; r < NR; ++r) {
      float t = Zs[l * NR + r] * d;
      float st = __sinf(t), ct = __cosf(t);
      float jl = st / t;
      if (l > 0) {
        float jm = jl;
        float jc = st / (t * t) - ct / t;
        for (int ll = 1; ll < l; ++ll) {
          float nx = (2.0f * ll + 1.0f) / t * jc - jm;
          jm = jc; jc = nx;
        }
        jl = jc;
      }
      srbf[(size_t)e * (NS * NR) + l * NR + r] = Norm[l * NR + r] * jl * env;
    }
  }
}

// ---------------------------------------------------------------------------
// Per-triplet: angle -> Legendre -> sbf, then fused [42]->[8] projections
// for both interaction blocks:  sb0[T,8], sb1[T,8]
// ---------------------------------------------------------------------------
__global__ void triplet_kernel(const float* __restrict__ R,
                               const int* __restrict__ t_i,
                               const int* __restrict__ t_j,
                               const int* __restrict__ t_k,
                               const int* __restrict__ id_expand,
                               const float* __restrict__ srbf,
                               const float* __restrict__ sbfW0,
                               const float* __restrict__ sbfW1,
                               float* __restrict__ sb0,
                               float* __restrict__ sb1, int T)
{
  int t = blockIdx.x * blockDim.x + threadIdx.x;
  if (t >= T) return;
  const int ia = t_i[t], ja = t_j[t], ka = t_k[t];
  float r1x = R[ja * 3 + 0] - R[ia * 3 + 0];
  float r1y = R[ja * 3 + 1] - R[ia * 3 + 1];
  float r1z = R[ja * 3 + 2] - R[ia * 3 + 2];
  float r2x = R[ka * 3 + 0] - R[ia * 3 + 0];
  float r2y = R[ka * 3 + 1] - R[ia * 3 + 1];
  float r2z = R[ka * 3 + 2] - R[ia * 3 + 2];
  float cx = r1y * r2z - r1z * r2y;
  float cy = r1z * r2x - r1x * r2z;
  float cz = r1x * r2y - r1y * r2x;
  float cn  = sqrtf(cx * cx + cy * cy + cz * cz);
  float dot = r1x * r2x + r1y * r2y + r1z * r2z;
  float ang = atan2f(cn, dot);
  float c   = __cosf(ang);

  float Pl[NS];
  Pl[0] = 1.0f; Pl[1] = c;
#pragma unroll
  for (int l = 1; l < NS - 1; ++l)
    Pl[l + 1] = ((2.0f * l + 1.0f) * c * Pl[l] - (float)l * Pl[l - 1]) / (float)(l + 1);

  const float inv4pi = 1.0f / (4.0f * 3.14159265358979323846f);
  const float* sr = srbf + (size_t)id_expand[t] * (NS * NR);
  float sv[NS * NR];
#pragma unroll
  for (int l = 0; l < NS; ++l) {
    float yc = sqrtf((2.0f * l + 1.0f) * inv4pi);
    float cb = yc * Pl[l];
#pragma unroll
    for (int r = 0; r < NR; ++r)
      sv[l * NR + r] = sr[l * NR + r] * cb;
  }
#pragma unroll
  for (int j = 0; j < NBIL; ++j) {
    float a0 = 0.0f, a1 = 0.0f;
#pragma unroll
    for (int q = 0; q < NS * NR; ++q) {
      a0 += sv[q] * sbfW0[q * NBIL + j];
      a1 += sv[q] * sbfW1[q * NBIL + j];
    }
    sb0[(size_t)t * NBIL + j] = a0;
    sb1[(size_t)t * NBIL + j] = a1;
  }
}

// ---------------------------------------------------------------------------
// K=6 projection  out = act(rbf @ W6x128 [+ bias]) [* x] ; store or atomic
// scatter to atoms via idnb_i (output-block segment_sum).
// ---------------------------------------------------------------------------
__global__ void rbfproj_kernel(const float* __restrict__ rbf,
                               const float* __restrict__ W,
                               const float* __restrict__ bias,
                               const float* __restrict__ xmul,
                               const int* __restrict__ scat_idx,
                               float* __restrict__ out, int E, int act)
{
  long gid = (long)blockIdx.x * blockDim.x + threadIdx.x;
  if (gid >= (long)E * 128) return;
  const int e = (int)(gid >> 7);
  const int f = (int)(gid & 127);
  const float* r = rbf + (size_t)e * NR;
  float v = 0.0f;
#pragma unroll
  for (int k = 0; k < NR; ++k) v += r[k] * W[k * 128 + f];
  if (bias) v += bias[f];
  if (act) v = swishf(v);
  if (xmul) v *= xmul[(size_t)e * 128 + f];
  if (scat_idx) atomicAdd(&out[(size_t)scat_idx[e] * 128 + f], v);
  else          out[(size_t)e * 128 + f] = v;
}

// ---------------------------------------------------------------------------
// Small utility kernels
// ---------------------------------------------------------------------------
__global__ void zero_kernel(float* __restrict__ p, long n) {
  long gid = (long)blockIdx.x * blockDim.x + threadIdx.x;
  if (gid < n) p[gid] = 0.0f;
}
__global__ void add_kernel(float* __restrict__ a, const float* __restrict__ b, long n) {
  long gid = (long)blockIdx.x * blockDim.x + threadIdx.x;
  if (gid < n) a[gid] += b[gid];
}
__global__ void compose_idx_kernel(const int* __restrict__ Z, const int* __restrict__ nb,
                                   int* __restrict__ out, int E) {
  int e = blockIdx.x * blockDim.x + threadIdx.x;
  if (e < E) out[e] = Z[nb[e]];
}
__global__ void seg_mol_kernel(const float* __restrict__ P, const int* __restrict__ bseg,
                               float* __restrict__ outm, int NA) {
  long gid = (long)blockIdx.x * blockDim.x + threadIdx.x;
  if (gid >= (long)NA * NTGT) return;
  const int a = (int)(gid / NTGT);
  const int f = (int)(gid % NTGT);
  atomicAdd(&outm[(size_t)bseg[a] * NTGT + f], P[(size_t)a * NTGT + f]);
}

// ===========================================================================
// Host orchestration
// ===========================================================================
// Input order: setup_inputs() dict, then params leaves in JAX pytree
// (sorted-dict) order.
enum {
  IN_Z = 0, IN_R, IN_BATCH, IN_NB_I, IN_NB_J, IN_EXP_KJ, IN_RED_JI,
  IN_T3_I, IN_T3_J, IN_T3_K,
  P_EMB_DENSE_W = 10, P_EMB_DENSE_B, P_EMB_RBF_W, P_EMB_RBF_B, P_EMB_TABLE,
  P_FREQ = 15
};
// int block bases (21 leaves each; keys sorted: W_bilin, after, before, final, ji, kj, rbf_W, sbf_W)
static const int INT_BASE[2] = { 16, 37 };
#define INT_WBILIN   0
#define INT_AFTER    1   // a0W1,a0b1,a0W2,a0b2,a1W1,a1b1,a1W2,a1b2
#define INT_BEFORE   9   // W1,b1,W2,b2
#define INT_FINAL_W 13
#define INT_FINAL_B 14
#define INT_JI_W    15
#define INT_JI_B    16
#define INT_KJ_W    17
#define INT_KJ_B    18
#define INT_RBF_W   19
#define INT_SBF_W   20
// out block bases (8 leaves each; keys sorted: dense, final_W, rbf_W)
static const int OUT_BASE[3] = { 58, 66, 74 };
#define OUT_D0W 0
#define OUT_D0B 1
#define OUT_D1W 2
#define OUT_D1B 3
#define OUT_D2W 4
#define OUT_D2B 5
#define OUT_FINW 6
#define OUT_RBFW 7

static inline int gemm_blocks(int M) { return (((M + 15) / 16) + 3) / 4; }

extern "C" void kernel_launch(void* const* d_in, const int* in_sizes, int n_in,
                              void* d_out, int out_size, void* d_ws, size_t ws_size,
                              hipStream_t stream)
{
  (void)n_in; (void)ws_size;
  const int NA = in_sizes[IN_Z];        // 15000 atoms
  const int E  = in_sizes[IN_NB_I];     // 120000 edges
  const int T  = in_sizes[IN_T3_I];     // 160000 triplets

  const float* R        = (const float*)d_in[IN_R];
  const int*   Zat      = (const int*)d_in[IN_Z];
  const int*   bseg     = (const int*)d_in[IN_BATCH];
  const int*   nb_i     = (const int*)d_in[IN_NB_I];
  const int*   nb_j     = (const int*)d_in[IN_NB_J];
  const int*   exp_kj   = (const int*)d_in[IN_EXP_KJ];
  const int*   red_ji   = (const int*)d_in[IN_RED_JI];
  const float* freq     = (const float*)d_in[P_FREQ];

  // ---- workspace carve-up (256B aligned) ----
  char* ws = (char*)d_ws;
  size_t cur = 0;
  auto alloc = [&](size_t bytes) -> char* {
    char* p = ws + cur;
    cur = (cur + bytes + 255) & ~(size_t)255;
    return p;
  };
  float* Zs    = (float*)alloc(42 * 4);
  float* Norm  = (float*)alloc(42 * 4);
  float* rbf   = (float*)alloc((size_t)E * NR * 4);
  float* srbf  = (float*)alloc((size_t)E * NS * NR * 4);
  float* sb0   = (float*)alloc((size_t)T * NBIL * 4);
  float* sb1   = (float*)alloc((size_t)T * NBIL * 4);
  int*   zi    = (int*)  alloc((size_t)E * 4);
  int*   zj    = (int*)  alloc((size_t)E * 4);
  float* B0    = (float*)alloc((size_t)E * 128 * 4);   // x
  float* B1    = (float*)alloc((size_t)E * 128 * 4);   // rbf_e / g / m
  float* B2    = (float*)alloc((size_t)E * 128 * 4);   // ji / x2
  float* B3    = (float*)alloc((size_t)E * 128 * 4);   // kj / tmp
  float* atomA = (float*)alloc((size_t)NA * 128 * 4);
  float* atomB = (float*)alloc((size_t)NA * 128 * 4);
  float* Pa    = (float*)alloc((size_t)NA * NTGT * 4);
  // packed bf16 weights: emb(49152) + 2*(17*16384) + 3*(3*16384 + 2048)
  bf16_t* pk   = (bf16_t*)alloc((size_t)(49152 + 2 * 17 * 16384 + 3 * (3 * 16384 + 2048)) * 2);

  // ---- pack all weights to bf16 WMMA-B layout ----
  size_t pkcur = 0;
  auto pack = [&](int leaf, int ksteps, int ntiles, int ncols,
                  long sK, long sN, long off) -> size_t {
    size_t at = pkcur;
    long total = (long)ksteps * ntiles * 512;
    pack_kernel<<<(int)((total + 255) / 256), 256, 0, stream>>>(
        (const float*)d_in[leaf], pk + at, ksteps, ntiles, ncols, sK, sN, off);
    pkcur += (size_t)total;
    return at;
  };

  basis_consts_kernel<<<1, 1, 0, stream>>>(Zs, Norm);

  const size_t pkEmb = pack(P_EMB_DENSE_W, 12, 8, 128, 128, 1, 0);
  size_t pkJi[2], pkKj[2], pkBW1[2], pkBW2[2], pkFin[2], pkA[2][4], pkBil[2];
  for (int ib = 0; ib < 2; ++ib) {
    const int b = INT_BASE[ib];
    pkJi[ib]  = pack(b + INT_JI_W,    4, 8, 128, 128, 1, 0);
    pkKj[ib]  = pack(b + INT_KJ_W,    4, 8, 128, 128, 1, 0);
    pkBW1[ib] = pack(b + INT_BEFORE + 0, 4, 8, 128, 128, 1, 0);
    pkBW2[ib] = pack(b + INT_BEFORE + 2, 4, 8, 128, 128, 1, 0);
    pkFin[ib] = pack(b + INT_FINAL_W, 4, 8, 128, 128, 1, 0);
    for (int r = 0; r < 4; ++r)  // a0W1,a0W2,a1W1,a1W2 at +1,+3,+5,+7
      pkA[ib][r] = pack(b + INT_AFTER + 2 * r, 4, 8, 128, 128, 1, 0);
    pkBil[ib] = pkcur;
    for (int j = 0; j < NBIL; ++j)  // M_j[k][n] = W_bilin[n, j, k]
      pack(b + INT_WBILIN, 4, 8, 128, /*sK*/1, /*sN*/(long)NBIL * 128, /*off*/(long)j * 128);
  }
  size_t pkOd[3][3], pkOf[3];
  for (int ob = 0; ob < 3; ++ob) {
    const int b = OUT_BASE[ob];
    pkOd[ob][0] = pack(b + OUT_D0W, 4, 8, 128, 128, 1, 0);
    pkOd[ob][1] = pack(b + OUT_D1W, 4, 8, 128, 128, 1, 0);
    pkOd[ob][2] = pack(b + OUT_D2W, 4, 8, 128, 128, 1, 0);
    pkOf[ob]    = pack(b + OUT_FINW, 4, 1, NTGT, NTGT, 1, 0);  // zero-pad N to 16
  }

  // ---- geometry / bases ----
  edge_geom_kernel<<<(E + 255) / 256, 256, 0, stream>>>(
      R, nb_i, nb_j, freq, Zs, Norm, rbf, srbf, E);
  triplet_kernel<<<(T + 255) / 256, 256, 0, stream>>>(
      R, (const int*)d_in[IN_T3_I], (const int*)d_in[IN_T3_J], (const int*)d_in[IN_T3_K],
      exp_kj, srbf,
      (const float*)d_in[INT_BASE[0] + INT_SBF_W],
      (const float*)d_in[INT_BASE[1] + INT_SBF_W],
      sb0, sb1, T);
  compose_idx_kernel<<<(E + 255) / 256, 256, 0, stream>>>(Zat, nb_i, zi, E);
  compose_idx_kernel<<<(E + 255) / 256, 256, 0, stream>>>(Zat, nb_j, zj, E);

  const long EF = (long)E * 128;
  const long AF = (long)NA * 128;
  const int  rbGrid = (int)((EF + 255) / 256);

  // rbf_e = swish(rbf @ emb.rbf.W + b)   -> B1
  rbfproj_kernel<<<rbGrid, 256, 0, stream>>>(
      rbf, (const float*)d_in[P_EMB_RBF_W], (const float*)d_in[P_EMB_RBF_B],
      nullptr, nullptr, B1, E, 1);
  // x = swish(concat[h_i, h_j, rbf_e] @ emb.dense.W + b)  -> B0
  wmma_gemm_kernel<3, 8><<<gemm_blocks(E), 128, 0, stream>>>(
      (const float*)d_in[P_EMB_TABLE], zi,
      (const float*)d_in[P_EMB_TABLE], zj,
      B1, nullptr,
      pk + pkEmb, (const float*)d_in[P_EMB_DENSE_B],
      nullptr, nullptr, B0, E, 128, 128, 1);

  // ---- output block helper ----
  auto out_block = [&](int ob, bool first) {
    zero_kernel<<<(int)((AF + 255) / 256), 256, 0, stream>>>(atomA, AF);
    rbfproj_kernel<<<rbGrid, 256, 0, stream>>>(
        rbf, (const float*)d_in[OUT_BASE[ob] + OUT_RBFW], nullptr,
        B0, nb_i, atomA, E, 0);
    float* src = atomA; float* dst = atomB;
    for (int dlayer = 0; dlayer < 3; ++dlayer) {
      wmma_gemm_kernel<1, 8><<<gemm_blocks(NA), 128, 0, stream>>>(
          src, nullptr, nullptr, nullptr, nullptr, nullptr,
          pk + pkOd[ob][dlayer], (const float*)d_in[OUT_BASE[ob] + OUT_D0B + 2 * dlayer],
          nullptr, nullptr, dst, NA, 128, 128, 1);
      float* t = src; src = dst; dst = t;
    }
    wmma_gemm_kernel<1, 1><<<gemm_blocks(NA), 128, 0, stream>>>(
        src, nullptr, nullptr, nullptr, nullptr, nullptr,
        pk + pkOf[ob], nullptr, nullptr, first ? nullptr : Pa,
        Pa, NA, NTGT, NTGT, 0);
  };

  out_block(0, true);

  // ---- interaction blocks ----
  for (int ib = 0; ib < 2; ++ib) {
    const int b = INT_BASE[ib];
    // g = rbf @ rbf_W  -> B1
    rbfproj_kernel<<<rbGrid, 256, 0, stream>>>(
        rbf, (const float*)d_in[b + INT_RBF_W], nullptr, nullptr, nullptr, B1, E, 0);
    // ji = swish(x @ Wji + b)  -> B2
    wmma_gemm_kernel<1, 8><<<gemm_blocks(E), 128, 0, stream>>>(
        B0, nullptr, nullptr, nullptr, nullptr, nullptr,
        pk + pkJi[ib], (const float*)d_in[b + INT_JI_B],
        nullptr, nullptr, B2, E, 128, 128, 1);
    // kj = swish(x @ Wkj + b) * g  -> B3
    wmma_gemm_kernel<1, 8><<<gemm_blocks(E), 128, 0, stream>>>(
        B0, nullptr, nullptr, nullptr, nullptr, nullptr,
        pk + pkKj[ib], (const float*)d_in[b + INT_KJ_B],
        B1, nullptr, B3, E, 128, 128, 1);
    // m = segment_sum_over_triplets(bilinear)  -> B1 (zeroed; g dead now)
    zero_kernel<<<(int)((EF + 255) / 256), 256, 0, stream>>>(B1, EF);
    bilinear_kernel<<<gemm_blocks(T), 128, 0, stream>>>(
        B3, exp_kj, (ib == 0) ? sb0 : sb1, pk + pkBil[ib], red_ji, B1, T);
    // x2 = ji + m   (B2 += B1)
    add_kernel<<<(int)((EF + 255) / 256), 256, 0, stream>>>(B2, B1, EF);
    // before residual: x2 += swish(swish(x2@W1+b1)@W2+b2)
    wmma_gemm_kernel<1, 8><<<gemm_blocks(E), 128, 0, stream>>>(
        B2, nullptr, nullptr, nullptr, nullptr, nullptr,
        pk + pkBW1[ib], (const float*)d_in[b + INT_BEFORE + 1],
        nullptr, nullptr, B3, E, 128, 128, 1);
    wmma_gemm_kernel<1, 8><<<gemm_blocks(E), 128, 0, stream>>>(
        B3, nullptr, nullptr, nullptr, nullptr, nullptr,
        pk + pkBW2[ib], (const float*)d_in[b + INT_BEFORE + 3],
        nullptr, B2, B2, E, 128, 128, 1);
    // x = x + swish(x2 @ Wf + bf)
    wmma_gemm_kernel<1, 8><<<gemm_blocks(E), 128, 0, stream>>>(
        B2, nullptr, nullptr, nullptr, nullptr, nullptr,
        pk + pkFin[ib], (const float*)d_in[b + INT_FINAL_B],
        nullptr, B0, B0, E, 128, 128, 1);
    // after residuals (x += swish(swish(x@W1+b1)@W2+b2)) x2
    for (int r = 0; r < 2; ++r) {
      wmma_gemm_kernel<1, 8><<<gemm_blocks(E), 128, 0, stream>>>(
          B0, nullptr, nullptr, nullptr, nullptr, nullptr,
          pk + pkA[ib][2 * r], (const float*)d_in[b + INT_AFTER + 4 * r + 1],
          nullptr, nullptr, B2, E, 128, 128, 1);
      wmma_gemm_kernel<1, 8><<<gemm_blocks(E), 128, 0, stream>>>(
          B2, nullptr, nullptr, nullptr, nullptr, nullptr,
          pk + pkA[ib][2 * r + 1], (const float*)d_in[b + INT_AFTER + 4 * r + 3],
          nullptr, B0, B0, E, 128, 128, 1);
    }
    out_block(ib + 1, false);
  }

  // ---- molecule segment-sum ----
  zero_kernel<<<(out_size + 255) / 256, 256, 0, stream>>>((float*)d_out, out_size);
  seg_mol_kernel<<<(int)(((long)NA * NTGT + 255) / 256), 256, 0, stream>>>(
      Pa, bseg, (float*)d_out, NA);
}
